// PositionMemoryUpdater_66864050864700
// MI455X (gfx1250) — matrix operand order repaired and checked by reading, not compiled, not run on hardware
//
#include <hip/hip_runtime.h>

typedef __attribute__((ext_vector_type(16))) __bf16 v16bf;
typedef __attribute__((ext_vector_type(8)))  float  v8f;
typedef unsigned short ushort_t;

#define MSG_STRIDE 272
#define MEM_STRIDE 188
#define MEM_DIM    172
#define MSG_DIM    256
#define KT_IH      8      // 256 / 32
#define KT_HH      6      // 192 / 32 (172 zero-padded)
#define NT_TOTAL   33     // 3 gates * 176 padded / 16
#define TILES_IH   (NT_TOTAL * KT_IH)   // 264
#define TILES_HH   (NT_TOTAL * KT_HH)   // 198
#define XSTRIDE    264    // bf16 per LDS row (256 + 8 pad) -> 4-bank row skew
#define HSTRIDE    200    // bf16 per LDS row (192 + 8 pad)
#define SSTRIDE    180    // f32 per staging row (176 + 4 pad)

__device__ __forceinline__ ushort_t f2bf(float f) {
    unsigned u = __float_as_uint(f);
    u = (u + 0x7FFFu + ((u >> 16) & 1u)) >> 16;   // round-to-nearest-even
    return (ushort_t)u;
}

// ---------------------------------------------------------------------------
// Pack W_ih / W_hh (f32 row-major) into bf16 tiles in a contiguous-K (same as A) WMMA B-operand
// register layout: tile = 32 lanes x 16 bf16; lane l -> column N = nt*16+(l&15);
// lane-half h=(l>>4) -> K = kt*32 + h*16 + e  (e = 0..15, contiguous).
// Per-gate N padded 172->176; K_hh padded 172->192 with zeros.
// ---------------------------------------------------------------------------
__global__ void pack_weights_kernel(const float* __restrict__ Wih,
                                    const float* __restrict__ Whh,
                                    ushort_t* __restrict__ packed) {
    int gid = blockIdx.x * blockDim.x + threadIdx.x;
    if (gid >= (TILES_IH + TILES_HH) * 32) return;
    int lane = gid & 31;
    int tile = gid >> 5;

    const float* W; int K, nt, kt;
    if (tile < TILES_IH) { W = Wih; K = MSG_DIM; nt = tile >> 3; kt = tile & 7; }
    else { int t = tile - TILES_IH; W = Whh; K = MEM_DIM; nt = t / KT_HH; kt = t - (t / KT_HH) * KT_HH; }

    int npk = nt * 16 + (lane & 15);     // padded N index in [0, 528)
    int g   = npk / 176;                 // gate 0..2
    int c   = npk - g * 176;             // in-gate channel, valid if < 172
    int kb  = kt * 32 + (lane >> 4) * 16;

    ushort_t out[16] __attribute__((aligned(16)));
#pragma unroll
    for (int e = 0; e < 16; e++) {
        int k = kb + e;
        float v = (c < MEM_DIM && k < K) ? W[(size_t)(g * MEM_DIM + c) * K + k] : 0.f;
        out[e] = f2bf(v);
    }
    uint4* dst = (uint4*)(packed + (size_t)tile * 512 + lane * 16);
    const uint4* s = (const uint4*)out;
    dst[0] = s[0];
    dst[1] = s[1];
}

// ---------------------------------------------------------------------------
__global__ void copy_f4_kernel(const float4* __restrict__ src,
                               float4* __restrict__ dst, int n) {
    int i = blockIdx.x * blockDim.x + threadIdx.x;
    int stride = gridDim.x * blockDim.x;
    for (; i < n; i += stride) dst[i] = src[i];
}

// ---------------------------------------------------------------------------
// K-loop: A operand from LDS (lane-half h holds K = 16h..16h+15 per 32-K step,
// matching packed B -> two ds_read_b128), B operand pre-packed in
// global (two global_load_b128). f32 accumulate.
// ---------------------------------------------------------------------------
__device__ __forceinline__ v8f mma_rows(const ushort_t* __restrict__ aLds, int astride,
                                        const ushort_t* __restrict__ bTiles, int nkt,
                                        int l16, int lh, int lane, v8f acc) {
    const ushort_t* ap0 = aLds + l16 * astride + lh * 16;
    const ushort_t* bp0 = bTiles + lane * 16;
    for (int kt = 0; kt < nkt; kt++) {
        union U { v16bf v; uint4 q[2]; };
        U A, B;
        A.q[0] = *(const uint4*)(ap0 + kt * 32);
        A.q[1] = *(const uint4*)(ap0 + kt * 32 + 8);
        B.q[0] = *(const uint4*)(bp0 + (size_t)kt * 512);
        B.q[1] = *(const uint4*)(bp0 + (size_t)kt * 512 + 8);
        acc = __builtin_amdgcn_wmma_f32_16x16x32_bf16(false, A.v, false, B.v,
                                                      (short)0, acc, false, false);
    }
    return acc;
}

// ---------------------------------------------------------------------------
// One block = 16 batch rows. Static LDS: 8448 + 6400 + 4*11520 = 60928 B.
// ---------------------------------------------------------------------------
__global__ __launch_bounds__(256)
void gru_update_kernel(const float* __restrict__ msgs,
                       const float* __restrict__ memory,
                       const float* __restrict__ timestamps,
                       const float* __restrict__ b_ih,
                       const float* __restrict__ b_hh,
                       const int* __restrict__ ids,
                       const ushort_t* __restrict__ packW,
                       float* __restrict__ out_mem,
                       float* __restrict__ out_last) {
    __shared__ __align__(16) ushort_t xlds[16 * XSTRIDE];
    __shared__ __align__(16) ushort_t hlds[16 * HSTRIDE];
    __shared__ float s_r [16 * SSTRIDE];   // gi_r + gh_r
    __shared__ float s_z [16 * SSTRIDE];   // gi_z + gh_z
    __shared__ float g_in[16 * SSTRIDE];   // gi_n
    __shared__ float g_hn[16 * SSTRIDE];   // gh_n

    const int tid = threadIdx.x;
    const int rowBase = blockIdx.x << 4;

    // Phase 1a: x tile -> LDS bf16 (16 rows x 256)
    {
        int r = tid >> 4, c0 = (tid & 15) << 4;
        const float* src = msgs + (size_t)(rowBase + r) * MSG_STRIDE + c0;
        ushort_t* dst = xlds + r * XSTRIDE + c0;
#pragma unroll
        for (int i = 0; i < 16; i += 4) {
            float4 v = *(const float4*)(src + i);
            dst[i + 0] = f2bf(v.x); dst[i + 1] = f2bf(v.y);
            dst[i + 2] = f2bf(v.z); dst[i + 3] = f2bf(v.w);
        }
    }
    // Phase 1b: h tile (gather by node id) -> LDS bf16, K padded to 192
    {
        int r = tid >> 4, c0 = (tid & 15) * 12;
        int id = ids[rowBase + r];
        const float* src = memory + (size_t)id * MEM_STRIDE;
        ushort_t* dst = hlds + r * HSTRIDE;
#pragma unroll
        for (int i = 0; i < 12; i++) {
            int c = c0 + i;
            dst[c] = (c < MEM_DIM) ? f2bf(src[c]) : (ushort_t)0;
        }
    }
    __syncthreads();

    // Phase 2: 44 tile-units over 8 waves. q=0: s_r (gi+gh fused accumulate),
    // q=1: s_z (fused), q=2: gi_n, q=3: gh_n. 462 WMMAs per block.
    const int wave = tid >> 5;
    const int lane = tid & 31;
    const int l16  = lane & 15;
    const int lh   = lane >> 4;
    const ushort_t* packIH = packW;
    const ushort_t* packHH = packW + (size_t)TILES_IH * 512;

    for (int u = wave; u < 44; u += 8) {
        int q   = u / 11;
        int ntg = u - q * 11;          // column tile within gate [0,11)
        v8f acc = {};
        if (q <= 1) {
            int ntAbs = q * 11 + ntg;
            acc = mma_rows(xlds, XSTRIDE, packIH + (size_t)(ntAbs * KT_IH) * 512, KT_IH, l16, lh, lane, acc);
            acc = mma_rows(hlds, HSTRIDE, packHH + (size_t)(ntAbs * KT_HH) * 512, KT_HH, l16, lh, lane, acc);
        } else if (q == 2) {
            int ntAbs = 22 + ntg;
            acc = mma_rows(xlds, XSTRIDE, packIH + (size_t)(ntAbs * KT_IH) * 512, KT_IH, l16, lh, lane, acc);
        } else {
            int ntAbs = 22 + ntg;
            acc = mma_rows(hlds, HSTRIDE, packHH + (size_t)(ntAbs * KT_HH) * 512, KT_HH, l16, lh, lane, acc);
        }
        // C/D layout: VGPR i holds row M = i + 8*lh, N = l16 (+ ntg*16)
        float* sdst = (q == 0 ? s_r : q == 1 ? s_z : q == 2 ? g_in : g_hn)
                      + (lh * 8) * SSTRIDE + ntg * 16 + l16;
#pragma unroll
        for (int i = 0; i < 8; i++) sdst[i * SSTRIDE] = acc[i];
    }
    __syncthreads();

    // Phase 3: GRU elementwise (f32) + scatter
    for (int e = tid; e < 16 * MEM_DIM; e += 256) {
        int m = e / MEM_DIM;
        int c = e - m * MEM_DIM;
        int id = ids[rowBase + m];
        float rr = s_r [m * SSTRIDE + c] + b_ih[c]               + b_hh[c];
        float zz = s_z [m * SSTRIDE + c] + b_ih[MEM_DIM + c]     + b_hh[MEM_DIM + c];
        float nn = g_in[m * SSTRIDE + c] + b_ih[2 * MEM_DIM + c];
        float hh = g_hn[m * SSTRIDE + c] + b_hh[2 * MEM_DIM + c];
        float r = 1.f / (1.f + __expf(-rr));
        float z = 1.f / (1.f + __expf(-zz));
        float na = nn + r * hh;
        float n  = 1.f - 2.f / (__expf(2.f * na) + 1.f);   // tanh(na)
        float h0 = memory[(size_t)id * MEM_STRIDE + c];
        out_mem[(size_t)id * MEM_STRIDE + c] = (1.f - z) * n + z * h0;
    }
    // Extra 16 cols carried from messages + last_update timestamps
    {
        int m = tid >> 4, j = tid & 15;
        int id = ids[rowBase + m];
        out_mem[(size_t)id * MEM_STRIDE + MEM_DIM + j] =
            msgs[(size_t)(rowBase + m) * MSG_STRIDE + MSG_DIM + j];
        if (j == 0) out_last[id] = timestamps[rowBase + m];
    }
}

// ---------------------------------------------------------------------------
extern "C" void kernel_launch(void* const* d_in, const int* in_sizes, int n_in,
                              void* d_out, int out_size, void* d_ws, size_t ws_size,
                              hipStream_t stream) {
    const float* msgs        = (const float*)d_in[0];   // 200000 x 272
    const float* memory      = (const float*)d_in[1];   // 500000 x 188
    const float* last_update = (const float*)d_in[2];   // 500000
    const float* timestamps  = (const float*)d_in[3];   // 200000
    const float* W_ih        = (const float*)d_in[4];   // 516 x 256
    const float* W_hh        = (const float*)d_in[5];   // 516 x 172
    const float* b_ih        = (const float*)d_in[6];   // 516
    const float* b_hh        = (const float*)d_in[7];   // 516
    const int*   ids         = (const int*)d_in[8];     // 200000

    float* out_mem  = (float*)d_out;                        // 500000*188
    float* out_last = out_mem + (size_t)500000 * MEM_STRIDE; // 500000
    ushort_t* packW = (ushort_t*)d_ws;                      // 462 KiB bf16 tiles

    // 1) bf16 weight pack (L2-resident, reused by all blocks)
    pack_weights_kernel<<<((TILES_IH + TILES_HH) * 32 + 255) / 256, 256, 0, stream>>>(
        W_ih, W_hh, packW);

    // 2) full-copy memory + last_update into outputs (overwritten rows come later)
    copy_f4_kernel<<<8192, 256, 0, stream>>>(
        (const float4*)memory, (float4*)out_mem, (500000 * MEM_STRIDE) / 4);
    copy_f4_kernel<<<512, 256, 0, stream>>>(
        (const float4*)last_update, (float4*)out_last, 500000 / 4);

    // 3) fused GRU update + scatter (12500 blocks x 16 rows)
    gru_update_kernel<<<200000 / 16, 256, 0, stream>>>(
        msgs, memory, timestamps, b_ih, b_hh, ids, packW, out_mem, out_last);
}
